// MoEModel_51608327029192
// MI455X (gfx1250) — matrix operand order repaired and checked by reading, not compiled, not run on hardware
//
#include <hip/hip_runtime.h>

typedef __attribute__((ext_vector_type(16))) _Float16 v16h;
typedef __attribute__((ext_vector_type(8)))  _Float16 h8;
typedef __attribute__((ext_vector_type(8)))  float    v8f;

#define WAVES_PER_BLOCK 8

__global__ __launch_bounds__(256) void moe_mixture_kernel(
    const float* __restrict__ ctx, const float* __restrict__ inp,
    const float* __restrict__ W1,  const float* __restrict__ B1,
    const float* __restrict__ W2,  const float* __restrict__ B2,
    float* __restrict__ out, int B, int nchunks)
{
    __shared__ __align__(16) _Float16 sW1h[256];
    __shared__ __align__(16) _Float16 sB1h[256];
    __shared__ __align__(16) _Float16 sW2h[256];
    __shared__ __align__(16) float    sB2[8];
    __shared__ __align__(16) float    sEC[WAVES_PER_BLOCK][128];   // [elem][expert]
    __shared__ __align__(16) float    sPIN[WAVES_PER_BLOCK][128];  // [elem][expert]

    const int tid = threadIdx.x;
    sW1h[tid] = (_Float16)W1[tid];
    sB1h[tid] = (_Float16)B1[tid];
    sW2h[tid] = (_Float16)W2[tid];
    if (tid < 8) sB2[tid] = B2[tid];
    __syncthreads();

    const int lane = tid & 31;
    const int n16  = lane & 15;       // A: row id, B/D: column id (expert)
    const int half = lane >> 4;
    const int ia0  = half;            // A k-run 0 (h8 index): {0..7}/{8..15}
    const int ia1  = 2 + half;        // A k-run 1 (h8 index): {16..23}/{24..31}
    const int ib0  = half;            // B k-run (h8 index): {0..7,16..23}/{8..15,24..31}
    const int wl   = tid >> 5;
    float* ecbuf = sEC[wl];
    float* pibuf = sPIN[wl];

    const float cini = (n16 < 8) ? sB2[n16] : 0.0f;
    const v8f  Cb    = { cini, cini, cini, cini, cini, cini, cini, cini };
    const v16h hzero = {};

    const int wave = wl + blockIdx.x * WAVES_PER_BLOCK;
    const int nwav = gridDim.x * WAVES_PER_BLOCK;

    for (int chunk = wave; chunk < nchunks; chunk += nwav) {
        const int b0 = chunk * 16;

        // x values for this wave's 5 tiles (4 context tiles + 1 input tile),
        // splatted to packed f16 for v_pk_fma_f16 layer-1.
        v16h xv[5];
#pragma unroll
        for (int t = 0; t < 5; ++t) {
            int rowA;
            float xf;
            if (t < 4) {
                rowA = b0 + 4 * t + (n16 >> 2);
                rowA = rowA < B ? rowA : B - 1;
                xf = ctx[rowA * 8 + 2 * (n16 & 3)];
            } else {
                rowA = b0 + n16; rowA = rowA < B ? rowA : B - 1;
                xf = inp[rowA];
            }
            const _Float16 xh = (_Float16)xf;
            v16h s;
#pragma unroll
            for (int j = 0; j < 16; ++j) s[j] = xh;
            xv[t] = s;
        }

        v8f acc[5] = { Cb, Cb, Cb, Cb, Cb };

        // Expert loop, unrolled x2 so consecutive iterations ping-pong A
        // registers: iteration e+1's packed-f16 builds co-execute with
        // iteration e's WMMAs with no WAR register overlap (no v_nop filler).
#pragma unroll 2
        for (int e = 0; e < 8; ++e) {
            const h8* w1p = (const h8*)(sW1h + e * 32);
            const h8* b1p = (const h8*)(sB1h + e * 32);
            const h8* w2p = (const h8*)(sW2h + e * 32);
            union { v16h v; h8 h[2]; } uw, ub, um;
            uw.h[0] = w1p[ia0];  uw.h[1] = w1p[ia1];
            ub.h[0] = b1p[ia0];  ub.h[1] = b1p[ia1];
            um.h[0] = w2p[ib0];  um.h[1] = w2p[ib0 + 2];
            const v16h wv = uw.v;
            const v16h bv = ub.v;
            const v16h bm = (n16 == e) ? um.v : hzero;  // block-column B(e)

            // Build all 5 A tiles first (distinct registers) ...
            v16h a[5];
#pragma unroll
            for (int t = 0; t < 5; ++t) {
                v16h aa = xv[t] * wv + bv;                   // v_pk_fma_f16 x8
                a[t] = __builtin_elementwise_max(aa, hzero); // v_pk_max_num_f16 x8
            }
            __builtin_amdgcn_sched_barrier(0);
            // ... then issue the 5 independent WMMAs back-to-back.
#pragma unroll
            for (int t = 0; t < 5; ++t)
                acc[t] = __builtin_amdgcn_wmma_f32_16x16x32_f16(
                    false, a[t], false, bm, (short)0, acc[t], false, false);
        }

        // ---- context errors: D vgpr r <-> point M = r + 8*half ----
#pragma unroll
        for (int t = 0; t < 4; ++t) {
            float ec0 = 0.0f, ec1 = 0.0f;
#pragma unroll
            for (int r = 0; r < 8; ++r) {
                int rowY = b0 + 4 * t + 2 * half + (r >> 2);
                rowY = rowY < B ? rowY : B - 1;
                const float y = ctx[rowY * 8 + 2 * (r & 3) + 1];
                const float d = acc[t][r] - y;
                if (r < 4) ec0 += d * d; else ec1 += d * d;
            }
            if (n16 < 8) {
                ecbuf[(4 * t + 2 * half)     * 8 + n16] = ec0;
                ecbuf[(4 * t + 2 * half + 1) * 8 + n16] = ec1;
            }
        }
        // ---- input-tile predictions ----
        if (n16 < 8) {
#pragma unroll
            for (int r = 0; r < 8; ++r)
                pibuf[(r + 8 * half) * 8 + n16] = acc[4][r];
        }

        // ---- softmax over experts + combine: one element per lane 0..15 ----
        // Intra-wave LDS: DS ops are in-order within a wave; no barrier needed.
        {
            const float4 ea = *(const float4*)(ecbuf + n16 * 8);
            const float4 eb = *(const float4*)(ecbuf + n16 * 8 + 4);
            const float4 pa = *(const float4*)(pibuf + n16 * 8);
            const float4 pb = *(const float4*)(pibuf + n16 * 8 + 4);

            float m = fminf(fminf(fminf(ea.x, ea.y), fminf(ea.z, ea.w)),
                            fminf(fminf(eb.x, eb.y), fminf(eb.z, eb.w)));
            const float w0 = __expf(m - ea.x), w1 = __expf(m - ea.y);
            const float w2 = __expf(m - ea.z), w3 = __expf(m - ea.w);
            const float w4 = __expf(m - eb.x), w5 = __expf(m - eb.y);
            const float w6 = __expf(m - eb.z), w7 = __expf(m - eb.w);
            const float s  = ((w0 + w1) + (w2 + w3)) + ((w4 + w5) + (w6 + w7));
            const float cb = ((w0 * pa.x + w1 * pa.y) + (w2 * pa.z + w3 * pa.w)) +
                             ((w4 * pb.x + w5 * pb.y) + (w6 * pb.z + w7 * pb.w));
            if (half == 0 && (b0 + n16) < B)
                out[b0 + n16] = cb / s;
        }
    }
}

extern "C" void kernel_launch(void* const* d_in, const int* in_sizes, int n_in,
                              void* d_out, int out_size, void* d_ws, size_t ws_size,
                              hipStream_t stream) {
    const float* ctx = (const float*)d_in[0];
    const float* inp = (const float*)d_in[1];
    const float* W1  = (const float*)d_in[2];
    const float* B1  = (const float*)d_in[3];
    const float* W2  = (const float*)d_in[4];
    const float* B2  = (const float*)d_in[5];
    float* out = (float*)d_out;

    const int B = out_size;                 // output is [B,1]
    const int nchunks = (B + 15) / 16;
    int blocks = (nchunks + WAVES_PER_BLOCK - 1) / WAVES_PER_BLOCK;
    if (blocks > 1024) blocks = 1024;
    if (blocks < 1)    blocks = 1;

    moe_mixture_kernel<<<blocks, 256, 0, stream>>>(ctx, inp, W1, B1, W2, B2,
                                                   out, B, nchunks);
}
